// SpikeNetEfficient_33105607917779
// MI455X (gfx1250) — matrix operand order repaired and checked by reading, not compiled, not run on hardware
//
#include <hip/hip_runtime.h>

typedef _Float16 v16h __attribute__((ext_vector_type(16)));
typedef _Float16 v8h  __attribute__((ext_vector_type(8)));
typedef float    v8f  __attribute__((ext_vector_type(8)));
typedef float    v4f  __attribute__((ext_vector_type(4)));
typedef unsigned int u32x4 __attribute__((ext_vector_type(4)));
typedef int          i32x8 __attribute__((ext_vector_type(8)));
typedef int          i32x4 __attribute__((ext_vector_type(4)));

#define T_   32
#define B_   8
#define L_   512
#define IN_  1611
#define KPAD 1632          // 51 * 32, zero-padded K for input GEMM
#define E_   256
#define C_   7
#define M1   (B_ * L_)     // 4096 (b,l) rows

__device__ __forceinline__ v8f wmma_f16(v16h a, v16h b, v8f c) {
  // D = A(16x32 f16) * B(32x16 f16) + C(16x16 f32)
  return __builtin_amdgcn_wmma_f32_16x16x32_f16(false, a, false, b, (short)0, c,
                                                false, false);
}

__device__ __forceinline__ v16h cat8(v8h lo, v8h hi) {
  return __builtin_shufflevector(lo, hi, 0, 1, 2, 3, 4, 5, 6, 7,
                                         8, 9, 10, 11, 12, 13, 14, 15);
}

// TDM: DMA 128KB (4096x16 tile of 2-byte elements, stride 4096) global -> LDS.
// D# packed per CDNA5 ISA 8.3/8.4.  Issue from ONE wave only (EXEC ignored).
// 6-arg builtin form (clang-23 / therock lane).
__device__ __forceinline__ void tdm_load_128k(unsigned lds_off,
                                              const _Float16* gsrc) {
  unsigned long long ga = (unsigned long long)(__UINTPTR_TYPE__)gsrc;
  u32x4 g0;
  g0[0] = 1u;                                  // count=1 (valid user D#)
  g0[1] = lds_off;                             // lds_addr [63:32]
  g0[2] = (unsigned)ga;                        // global_addr lo
  g0[3] = (unsigned)(ga >> 32) | (2u << 30);   // global_addr[56:32] | type=2
  i32x8 g1;
  g1[0] = (int)(1u << 16);          // workgroup_mask=0, data_size=1 (2 bytes)
  g1[1] = (int)(4096u << 16);       // tensor_dim0=4096 (low16 @ bits 63:48)
  g1[2] = (int)(16u << 16);         // tensor_dim0 hi=0 | tensor_dim1=16 lo16
  g1[3] = (int)(4096u << 16);       // tensor_dim1 hi=0 | tile_dim0=4096
  g1[4] = 16;                       // tile_dim1=16, tile_dim2=0
  g1[5] = 4096;                     // tensor_dim0_stride=4096 (lo32)
  g1[6] = 0;                        // stride hi | tensor_dim1_stride lo
  g1[7] = 0;
  i32x4 z4 = {0, 0, 0, 0};          // group2/group3: unused (2D tensor)
  i32x8 z8 = {0, 0, 0, 0, 0, 0, 0, 0};
  __builtin_amdgcn_tensor_load_to_lds(g0, g1, z4, z4, z8, 0);
}

// ---------------------------------------------------------------------------
// Kernel 1: convert weights to f16, transposed [N][K] so B-fragments are
// contiguous 128-bit loads; pad W_in's K dim 1611 -> 1632 with zeros.
// ---------------------------------------------------------------------------
__global__ void prep_kernel(const float* __restrict__ W_in,
                            const float* __restrict__ W1,
                            const float* __restrict__ W2,
                            _Float16* __restrict__ Winh,
                            _Float16* __restrict__ W1h,
                            _Float16* __restrict__ W2h) {
  const int total = E_ * KPAD + 2 * E_ * E_;
  for (int i = blockIdx.x * blockDim.x + threadIdx.x; i < total;
       i += gridDim.x * blockDim.x) {
    if (i < E_ * KPAD) {
      int n = i / KPAD, k = i % KPAD;
      Winh[i] = (k < IN_) ? (_Float16)W_in[k * E_ + n] : (_Float16)0.0f;
    } else if (i < E_ * KPAD + E_ * E_) {
      int j = i - E_ * KPAD;
      int n = j / E_, k = j % E_;
      W1h[j] = (_Float16)W1[k * E_ + n];
    } else {
      int j = i - E_ * KPAD - E_ * E_;
      int n = j / E_, k = j % E_;
      W2h[j] = (_Float16)W2[k * E_ + n];
    }
  }
}

// ---------------------------------------------------------------------------
// Kernel 2: p = sigmoid(x @ W_in).  M=4096, K=1611(->1632), N=256.
// One wave per 16x16 tile, 4 waves / block. grid = (256 Mtiles, 4), block 128.
// ---------------------------------------------------------------------------
__global__ void gemm_in_kernel(const float* __restrict__ x,
                               const _Float16* __restrict__ Winh,
                               float* __restrict__ p) {
  const int lane  = threadIdx.x & 31;
  const int wv    = threadIdx.x >> 5;           // 0..3
  const int mtile = blockIdx.x;                 // 0..255
  const int ntile = blockIdx.y * 4 + wv;        // 0..15
  const int hi    = lane >> 4;                  // lane-half select
  const int rowA  = mtile * 16 + (lane & 15);   // global M row
  const int nB    = ntile * 16 + (lane & 15);   // global N col

  const float* xrow = x + (long)rowA * IN_;
  const _Float16* brow = Winh + (long)nB * KPAD;

  v8f acc = {};
  for (int kt = 0; kt < KPAD / 32; ++kt) {
    const int k0 = kt * 32 + hi * 8;
    v16h a;
#pragma unroll
    for (int i = 0; i < 8; ++i) {
      const int ka = k0 + i;           // K = base + {0..7}
      const int kb = k0 + 16 + i;      // K = base + {16..23}
      a[i]     = (ka < IN_) ? (_Float16)xrow[ka] : (_Float16)0.0f;
      a[i + 8] = (kb < IN_) ? (_Float16)xrow[kb] : (_Float16)0.0f;
    }
    const v8h blo = *(const v8h*)(brow + kt * 32 + hi * 8);
    const v8h bhi = *(const v8h*)(brow + kt * 32 + hi * 8 + 16);
    acc = wmma_f16(a, cat8(blo, bhi), acc);
  }
#pragma unroll
  for (int r = 0; r < 8; ++r) {
    const int row = mtile * 16 + r + hi * 8;
    const int col = ntile * 16 + (lane & 15);
    const float h = acc[r];
    p[row * E_ + col] = 1.0f / (1.0f + __expf(-h));
  }
}

// ---------------------------------------------------------------------------
// Kernel 3: fused rate-encode + LIF1 + (s1@W1) + LIF2 + (s2@W2) + LIF3 + mean.
// WG g owns 16 consecutive (b,l) rows (all same b). 512 threads = 16 waves.
// Weights staged once into LDS by the Tensor Data Mover; spike tile in LDS;
// LIF states in registers; u[t+1] software-prefetched across the GEMMs.
// ---------------------------------------------------------------------------
__global__ void __launch_bounds__(512) fused_kernel(
    const float* __restrict__ p, const float* __restrict__ u,
    const _Float16* __restrict__ W1h, const _Float16* __restrict__ W2h,
    float* __restrict__ partial) {
  __shared__ __align__(16) _Float16 w1_lds[E_ * E_];   // 128 KB
  __shared__ __align__(16) _Float16 w2_lds[E_ * E_];   // 128 KB
  __shared__ __align__(16) _Float16 s_s[16 * 264];     // spike tile, padded

  const int g    = blockIdx.x;        // 0..255
  const int tid  = threadIdx.x;
  const int lane = tid & 31;
  const int wv   = tid >> 5;          // wave id 0..15 == elementwise row
  const int hi   = lane >> 4;

  // Wave 0 kicks off the TDM weight DMA; overlaps with p loads below.
  if (wv == 0) {
    tdm_load_128k((unsigned)(__UINTPTR_TYPE__)&w1_lds[0], W1h);
    tdm_load_128k((unsigned)(__UINTPTR_TYPE__)&w2_lds[0], W2h);
  }

  // elementwise mapping: 8 elements per thread, row = wv, cols = lane*8..+7
  const int erow = wv;
  const int ecol = lane * 8;
  const long Rrow = (long)g * 16 + erow;        // global (b,l) row

  float pp[8], v1[8];
#pragma unroll
  for (int j = 0; j < 8; ++j) {
    pp[j] = p[Rrow * E_ + ecol + j];
    v1[j] = 0.0f;
  }

  // per-wave GEMM constants: wave wv owns output cols [wv*16, wv*16+16)
  const int rowA = lane & 15;
  const int nloc = wv * 16 + (lane & 15);
  const _Float16* b1row = w1_lds + nloc * E_;
  const _Float16* b2row = w2_lds + nloc * E_;

  // prefetch u[t=0]
  const float* ubase = u + Rrow * E_ + ecol;
  v4f u0 = *(const v4f*)(ubase);
  v4f u1 = *(const v4f*)(ubase + 4);

  if (wv == 0) __builtin_amdgcn_s_wait_tensorcnt(0);
  __syncthreads();                    // weights visible to all waves

  v8f v2 = {}, v3 = {}, sacc = {};

  for (int t = 0; t < T_; ++t) {
    // ---- Bernoulli encode + LIF layer 1 (elementwise, fp32 state) ----
#pragma unroll
    for (int j = 0; j < 8; ++j) {
      const float uv = (j < 4) ? u0[j] : u1[j - 4];
      const float sin_ = (uv <= pp[j]) ? 1.0f : 0.0f;
      const float v = v1[j] + (sin_ - v1[j]) * 0.5f;   // tau = 2
      const float s = (v >= 1.0f) ? 1.0f : 0.0f;       // v_th = 1
      v1[j] = (1.0f - s) * v;                          // hard reset
      s_s[erow * 264 + ecol + j] = (_Float16)s;
    }
    __syncthreads();

    // software-prefetch u[t+1]; flies during both GEMMs
    if (t < T_ - 1) {
      const float* un = ubase + (long)(t + 1) * M1 * E_;
      u0 = *(const v4f*)(un);
      u1 = *(const v4f*)(un + 4);
    }

    // ---- GEMM1: y1 = s1 @ W1 (A and B both from LDS) ----
    v8f y1 = {};
#pragma unroll
    for (int kt = 0; kt < 8; ++kt) {
      const int k0 = kt * 32 + hi * 8;
      const v8h alo = *(const v8h*)(&s_s[rowA * 264 + k0]);
      const v8h ahi = *(const v8h*)(&s_s[rowA * 264 + k0 + 16]);
      const v8h blo = *(const v8h*)(b1row + kt * 32 + hi * 8);
      const v8h bhi = *(const v8h*)(b1row + kt * 32 + hi * 8 + 16);
      y1 = wmma_f16(cat8(alo, ahi), cat8(blo, bhi), y1);
    }
    __syncthreads();   // all waves done reading s1 before overwrite

    // ---- LIF layer 2 (state in accumulator registers), write s2 tile ----
#pragma unroll
    for (int r = 0; r < 8; ++r) {
      const float v = v2[r] + (y1[r] - v2[r]) * 0.5f;
      const float s = (v >= 1.0f) ? 1.0f : 0.0f;
      v2[r] = (1.0f - s) * v;
      s_s[(r + hi * 8) * 264 + nloc] = (_Float16)s;   // C-layout scatter
    }
    __syncthreads();

    // ---- GEMM2: y2 = s2 @ W2 ----
    v8f y2 = {};
#pragma unroll
    for (int kt = 0; kt < 8; ++kt) {
      const int k0 = kt * 32 + hi * 8;
      const v8h alo = *(const v8h*)(&s_s[rowA * 264 + k0]);
      const v8h ahi = *(const v8h*)(&s_s[rowA * 264 + k0 + 16]);
      const v8h blo = *(const v8h*)(b2row + kt * 32 + hi * 8);
      const v8h bhi = *(const v8h*)(b2row + kt * 32 + hi * 8 + 16);
      y2 = wmma_f16(cat8(alo, ahi), cat8(blo, bhi), y2);
    }

    // ---- LIF layer 3 + accumulate spike sums (mean over T,L later) ----
#pragma unroll
    for (int r = 0; r < 8; ++r) {
      const float v = v3[r] + (y2[r] - v3[r]) * 0.5f;
      const float s = (v >= 1.0f) ? 1.0f : 0.0f;
      v3[r] = (1.0f - s) * v;
      sacc[r] += s;
    }
    __syncthreads();   // s_s free for next t's s1
  }

  // reduce over the 16 local rows (all same b): r-sum, then half-wave swap
  float tot = 0.0f;
#pragma unroll
  for (int r = 0; r < 8; ++r) tot += sacc[r];
  tot += __shfl_xor(tot, 16, 32);      // add M=r and M=r+8 halves (same N)
  if (lane < 16) partial[g * E_ + wv * 16 + lane] = tot;  // deterministic
}

// ---------------------------------------------------------------------------
// Kernel 4: deterministic reduction of per-WG partials + head GEMM (8x7).
// feat[b,e] = sum_g partial / (T*L);  out = feat @ Wc + bc
// ---------------------------------------------------------------------------
__global__ void head_kernel(const float* __restrict__ partial,
                            const float* __restrict__ Wc,
                            const float* __restrict__ bc,
                            float* __restrict__ out) {
  const int t = threadIdx.x;
  if (t >= B_ * C_) return;
  const int b = t / C_, c = t % C_;
  float acc = 0.0f;
  for (int e = 0; e < E_; ++e) {
    float s = 0.0f;
    for (int gg = 0; gg < 32; ++gg)          // 32 WGs per batch b
      s += partial[(b * 32 + gg) * E_ + e];
    acc += s * Wc[e * C_ + c];
  }
  out[t] = acc * (1.0f / (T_ * L_)) + bc[c];
}

// ---------------------------------------------------------------------------
extern "C" void kernel_launch(void* const* d_in, const int* in_sizes, int n_in,
                              void* d_out, int out_size, void* d_ws,
                              size_t ws_size, hipStream_t stream) {
  const float* x    = (const float*)d_in[0];
  const float* u    = (const float*)d_in[1];
  const float* W_in = (const float*)d_in[2];
  const float* W1   = (const float*)d_in[3];
  const float* W2   = (const float*)d_in[4];
  const float* Wc   = (const float*)d_in[5];
  const float* bc   = (const float*)d_in[6];
  float* out = (float*)d_out;

  char* ws = (char*)d_ws;
  float*    p       = (float*)(ws);                        // 4096*256*4 = 4,194,304 B
  _Float16* Winh    = (_Float16*)(ws + 4194304);           // 256*1632*2 =   835,584 B
  _Float16* W1h     = (_Float16*)(ws + 4194304 + 835584);  //               131,072 B
  _Float16* W2h     = (_Float16*)(ws + 4194304 + 835584 + 131072);
  float*    partial = (float*)(ws + 4194304 + 835584 + 262144);  // 256*256*4

  hipLaunchKernelGGL(prep_kernel, dim3(512), dim3(256), 0, stream,
                     W_in, W1, W2, Winh, W1h, W2h);
  hipLaunchKernelGGL(gemm_in_kernel, dim3(256, 4), dim3(128), 0, stream,
                     x, Winh, p);
  hipLaunchKernelGGL(fused_kernel, dim3(256), dim3(512), 0, stream,
                     p, u, W1h, W2h, partial);
  hipLaunchKernelGGL(head_kernel, dim3(1), dim3(64), 0, stream,
                     partial, Wc, bc, out);
}
